// BatchColumnShardedLinear_48885317763196
// MI455X (gfx1250) — matrix operand order repaired and checked by reading, not compiled, not run on hardware
//
#include <hip/hip_runtime.h>
#include <hip/hip_bf16.h>
#include <stdint.h>

#define G_EXPERTS 8
#define T_TOKENS  16384
#define D_IN      2048
#define D_OUT     8192

#define BM 64
#define BN 128
#define BK 32
#define LDSROW 20   // dwords per LDS row: 16 used + 4 pad (keeps b128 16B-aligned)
#define NKSTEPS (D_IN / BK)

typedef __bf16 bf16_t;
typedef bf16_t v16bf __attribute__((ext_vector_type(16)));
typedef float  v8f   __attribute__((ext_vector_type(8)));

union Frag { uint4 q[2]; v16bf v; };

// Split two fp32 values into packed bf16 hi parts and packed bf16 lo (residual) parts.
// hi = top 16 bits (truncation); residual r = a - hi is exact; lo = top 16 bits of r.
// Dropped lo*lo term ~ 2^-16 relative -> fp32-class accuracy from 3-term WMMA.
__device__ __forceinline__ void split_pair(float a, float b, uint32_t& hi, uint32_t& lo) {
  uint32_t ua = __float_as_uint(a), ub = __float_as_uint(b);
  uint32_t ha = ua & 0xffff0000u,   hb = ub & 0xffff0000u;
  float ra = a - __uint_as_float(ha);
  float rb = b - __uint_as_float(hb);
  hi = (ha >> 16) | hb;                                              // [15:0]=bf16(a), [31:16]=bf16(b)
  lo = (__float_as_uint(ra) >> 16) | (__float_as_uint(rb) & 0xffff0000u);
}

__global__ __launch_bounds__(256)
void grouped_gemm_bf16x3(const float* __restrict__ x, const float* __restrict__ w,
                         const int* __restrict__ counts, float* __restrict__ out) {
  // double-buffered staging: 2 x ~30KB (320KB/WGP available)
  __shared__ __align__(16) uint32_t As_hi[2][BM * LDSROW];
  __shared__ __align__(16) uint32_t As_lo[2][BM * LDSROW];
  __shared__ __align__(16) uint32_t Bs_hi[2][BN * LDSROW];
  __shared__ __align__(16) uint32_t Bs_lo[2][BN * LDSROW];

  // ---- resolve expert + row range for this padded row-tile (uniform scalar work) ----
  const int rowtile = blockIdx.y;
  int gsel = -1, row_start = 0, gend = 0;
  {
    int tileacc = 0, gstart = 0;
    for (int g = 0; g < G_EXPERTS; ++g) {
      int n = counts[g];
      int tg = (n + BM - 1) >> 6;
      if (gsel < 0 && rowtile < tileacc + tg) {
        gsel = g;
        row_start = gstart + (rowtile - tileacc) * BM;
        gend = gstart + n;
      }
      tileacc += tg;
      gstart  += n;
    }
  }
  if (gsel < 0) return;  // grid.y is an upper bound on the padded schedule

  const float* W    = w + (size_t)gsel * D_IN * D_OUT;
  const int    col0 = blockIdx.x * BN;

  const int tid    = threadIdx.x;
  const int wave   = tid >> 5;
  const int lane   = tid & 31;
  const int m_sub  = wave & 3;   // 4 M-subtiles of 16 rows
  const int n_half = wave >> 2;  // 2 N-halves of 64 cols
  const int lrow   = lane & 15;
  const int half   = lane >> 4;

  v8f acc[4] = {};

  // ---- staging assignments ----
  // A: thread -> (row, 8 consecutive k); 2 x global b128 per step
  const int a_row = tid >> 2;         // 0..63
  const int a_k   = (tid & 3) * 8;    // 0,8,16,24
  // B: thread -> (4 consecutive k-rows, 4 consecutive n); 4 x global b128 per step
  const int b_n4  = (tid & 31) * 4;   // 0..124
  const int b_r0  = (tid >> 5) * 4;   // k-row base 0..28
  const int b_p0  = b_r0 >> 1;        // k-pair base (even)

  int a_grow = row_start + a_row;     // clamp padded rows: still valid memory, masked at store
  if (a_grow > T_TOKENS - 1) a_grow = T_TOKENS - 1;
  const float* aptr = x + (size_t)a_grow * D_IN + a_k;
  const float* bptr = W + (size_t)b_r0 * D_OUT + col0 + b_n4;

  // staged global data (registers)
  float4 av0, av1, bv0, bv1, bv2, bv3;

  // issue global loads for K-step k0
  auto gload = [&](int k0) {
    av0 = *(const float4*)(aptr + k0);
    av1 = *(const float4*)(aptr + k0 + 4);
    const float* bp = bptr + (size_t)k0 * D_OUT;
    bv0 = *(const float4*)(bp);
    bv1 = *(const float4*)(bp + D_OUT);
    bv2 = *(const float4*)(bp + 2 * (size_t)D_OUT);
    bv3 = *(const float4*)(bp + 3 * (size_t)D_OUT);
  };

  gload(0);
  int cur = 0;

  for (int kt = 0; kt < NKSTEPS; ++kt) {
    // ---- convert staged registers -> LDS[cur] ----
    {
      uint32_t h0, l0, h1, l1, h2, l2, h3, l3;
      split_pair(av0.x, av0.y, h0, l0);
      split_pair(av0.z, av0.w, h1, l1);
      split_pair(av1.x, av1.y, h2, l2);
      split_pair(av1.z, av1.w, h3, l3);
      const int aoff = a_row * LDSROW + (a_k >> 1);
      *(uint4*)&As_hi[cur][aoff] = make_uint4(h0, h1, h2, h3);
      *(uint4*)&As_lo[cur][aoff] = make_uint4(l0, l1, l2, l3);

      // B transpose-pack: column n gets pair p0 = {rows r0,r0+1}, p0+1 = {rows r0+2,r0+3}
      const float c0[4] = {bv0.x, bv0.y, bv0.z, bv0.w};
      const float c1[4] = {bv1.x, bv1.y, bv1.z, bv1.w};
      const float c2[4] = {bv2.x, bv2.y, bv2.z, bv2.w};
      const float c3[4] = {bv3.x, bv3.y, bv3.z, bv3.w};
#pragma unroll
      for (int c = 0; c < 4; ++c) {
        uint32_t hA, lA, hB, lB;
        split_pair(c0[c], c1[c], hA, lA);
        split_pair(c2[c], c3[c], hB, lB);
        const int boff = (b_n4 + c) * LDSROW + b_p0;   // 8B-aligned (b_p0 even)
        *(uint2*)&Bs_hi[cur][boff] = make_uint2(hA, hB);
        *(uint2*)&Bs_lo[cur][boff] = make_uint2(lA, lB);
      }
    }

    // ---- prefetch next K-step's globals while WMMA phase runs ----
    if (kt + 1 < NKSTEPS) gload((kt + 1) * BK);

    __syncthreads();   // single barrier per K-step (double-buffered)

    // ---- A fragments (16x32 bf16: lanes 0-15 K{0..7,16..23}, lanes 16-31 K{8..15,24..31}) ----
    Frag a_hi, a_lo;
    const int ar = (m_sub * 16 + lrow) * LDSROW;
    a_hi.q[0] = *(const uint4*)&As_hi[cur][ar + half * 4];
    a_hi.q[1] = *(const uint4*)&As_hi[cur][ar + 8 + half * 4];
    a_lo.q[0] = *(const uint4*)&As_lo[cur][ar + half * 4];
    a_lo.q[1] = *(const uint4*)&As_lo[cur][ar + 8 + half * 4];

#pragma unroll
    for (int ns = 0; ns < 4; ++ns) {
      // B fragment (32x16 bf16: lanes 0-15 col N K{0..7,16..23}, lanes 16-31 K{8..15,24..31})
      Frag b_hi, b_lo;
      const int br = (n_half * 64 + ns * 16 + lrow) * LDSROW + half * 4;
      b_hi.q[0] = *(const uint4*)&Bs_hi[cur][br];
      b_hi.q[1] = *(const uint4*)&Bs_hi[cur][br + 8];
      b_lo.q[0] = *(const uint4*)&Bs_lo[cur][br];
      b_lo.q[1] = *(const uint4*)&Bs_lo[cur][br + 8];
      // 3-term fp32-emulating accumulation: hi*hi + hi*lo + lo*hi
      acc[ns] = __builtin_amdgcn_wmma_f32_16x16x32_bf16(false, a_hi.v, false, b_hi.v,
                                                        (short)0, acc[ns], false, false);
      acc[ns] = __builtin_amdgcn_wmma_f32_16x16x32_bf16(false, a_hi.v, false, b_lo.v,
                                                        (short)0, acc[ns], false, false);
      acc[ns] = __builtin_amdgcn_wmma_f32_16x16x32_bf16(false, a_lo.v, false, b_hi.v,
                                                        (short)0, acc[ns], false, false);
    }

    cur ^= 1;
  }

  // ---- store (mask ragged group boundary; C/D layout: VGPR i -> M = half*8 + i, N = lrow) ----
  const int out_col_base = col0 + n_half * 64;
#pragma unroll
  for (int ns = 0; ns < 4; ++ns) {
    const int col = out_col_base + ns * 16 + lrow;
#pragma unroll
    for (int i = 0; i < 8; ++i) {
      const int row = row_start + m_sub * 16 + half * 8 + i;
      if (row < gend) out[(size_t)row * D_OUT + col] = acc[ns][i];
    }
  }
}

extern "C" void kernel_launch(void* const* d_in, const int* in_sizes, int n_in,
                              void* d_out, int out_size, void* d_ws, size_t ws_size,
                              hipStream_t stream) {
  const float* x      = (const float*)d_in[0];
  const float* w      = (const float*)d_in[1];
  const int*   counts = (const int*)d_in[2];
  float*       out    = (float*)d_out;

  // grid.y is an upper bound on per-group padded row tiles: T/BM + (G-1) rounded up
  dim3 grid(D_OUT / BN, T_TOKENS / BM + G_EXPERTS, 1);
  dim3 block(256, 1, 1);
  hipLaunchKernelGGL(grouped_gemm_bf16x3, grid, block, 0, stream, x, w, counts, out);
}
